// GIN_edge_61761629716807
// MI455X (gfx1250) — matrix operand order repaired and checked by reading, not compiled, not run on hardware
//
#include <hip/hip_runtime.h>
#include <hip/hip_bf16.h>

// ---------------------------------------------------------------------------
// GINE (2-layer) + mean-pool + FC for MI455X (gfx1250, wave32, WMMA).
// Node MLP 64x64 GEMMs run on v_wmma_f32_16x16x32_f16 (fp32 accum).
// Weight tiles staged TRANSPOSED in LDS so B fragments load as ds_load_b128.
// Outputs staged in LDS and stored coalesced (b128) with a uniform tail bound.
// Edge scatter uses native f32 global atomics (unsafeAtomicAdd).
// ---------------------------------------------------------------------------

#define N_NODES  50000
#define N_EDGES  1600000
#define N_GRAPHS 1000

typedef __attribute__((ext_vector_type(16))) _Float16 v16h;
typedef __attribute__((ext_vector_type(8)))  _Float16 v8h;
typedef __attribute__((ext_vector_type(8)))  float    v8f;

__device__ __forceinline__ void atomicAddF(float* p, float v) {
  unsafeAtomicAdd(p, v);   // native global_atomic_add_f32
}

// ---- WMMA fragment loaders (contiguous K order in A and B, wave32) --------------------
// A (16x32 f16, MxK) from row-major [*][64] f16 tile:
//   lanes 0-15 row m=lane hold K 0..15; lanes 16-31 hold K 16..31.
// Both 8-half groups are 16B-aligned contiguous -> ds_load_b128 x2.
__device__ __forceinline__ v16h load_A_frag(const _Float16* Z,
                                            int row0, int k0, int lane) {
  int m  = row0 + (lane & 15);
  int kh = (lane >> 4) << 4;           // 0 or 16
  const _Float16* p = Z + m * 64 + k0 + kh;
  v8h lo = *(const v8h*)p;
  v8h hi = *(const v8h*)(p + 8);
  v16h a;
#pragma unroll
  for (int i = 0; i < 8; ++i) { a[i] = lo[i]; a[8 + i] = hi[i]; }
  return a;
}
// B (32x16 f16, KxN) from TRANSPOSED weight tile Wt[n][k] (64x64 f16):
//   lanes 0-15 col n=lane hold K 0..15; lanes 16-31 col n=lane-16 hold K 16..31.
//   16 K-contiguous halves = 32 contiguous bytes -> ds_load_b128 x2.
__device__ __forceinline__ v16h load_B_fragT(const _Float16* Wt,
                                             int k0, int n0, int lane) {
  int n  = n0 + (lane & 15);
  int kb = (lane >> 4) << 4;           // 0 or 16
  const _Float16* p = Wt + n * 64 + k0 + kb;
  v8h lo = *(const v8h*)p;
  v8h hi = *(const v8h*)(p + 8);
  v16h b;
#pragma unroll
  for (int i = 0; i < 8; ++i) { b[i] = lo[i]; b[8 + i] = hi[i]; }
  return b;
}

// ---------------------------------------------------------------------------
__global__ __launch_bounds__(256) void k_zero(float* __restrict__ p, int n) {
  int i = blockIdx.x * 256 + threadIdx.x;
  if (i < n) p[i] = 0.0f;
}

// ---- Layer 1 edge phase: aggr1[dst] += relu(x[src] + edge_attr@We1 + be1) --
__global__ __launch_bounds__(256) void k_edge1(
    const float* __restrict__ edge_attr, const int* __restrict__ eidx,
    const float* __restrict__ x, const float* __restrict__ We1,
    const float* __restrict__ be1, float* __restrict__ aggr1) {
  __shared__ float we[28];
  __shared__ float be[7];
  int tid = threadIdx.x;
  if (tid < 28) we[tid] = We1[tid];
  if (tid < 7)  be[tid] = be1[tid];
  __syncthreads();
  int e = blockIdx.x * 256 + tid;
  if (e >= N_EDGES) return;
  float4 ea = ((const float4*)edge_attr)[e];
  int src = eidx[e];
  int dst = eidx[N_EDGES + e];
#pragma unroll
  for (int d = 0; d < 7; ++d) {
    float v = be[d] + ea.x * we[d] + ea.y * we[7 + d]
                    + ea.z * we[14 + d] + ea.w * we[21 + d];
    v += x[src * 7 + d];
    v = fmaxf(v, 0.0f);
    atomicAddF(aggr1 + dst * 7 + d, v);
  }
}

// ---- Layer 1 node MLP: h1 = relu( relu((x+aggr1)@W1a+b1a) @ W1b + b1b ) ----
__global__ __launch_bounds__(256) void k_node_mlp1(
    const float* __restrict__ x, const float* __restrict__ aggr1,
    const float* __restrict__ W1a, const float* __restrict__ b1a,
    const float* __restrict__ W1b, const float* __restrict__ b1b,
    float* __restrict__ h1) {
  __shared__ float    zs[128][8];        // z rows (7 used)
  __shared__ float    w1a_s[7][64];
  __shared__ float    b1a_s[64];
  __shared__ _Float16 t16[128][64];      // relu(z@W1a+b1a) as f16 (A tiles)
  __shared__ _Float16 w1bt[64][64];      // W1b transposed: w1bt[n][k]
  __shared__ float    outs[128][64];     // f32 result staging
  int tid = threadIdx.x;
  int nodeBase = blockIdx.x * 128;
  for (int i = tid; i < 128 * 7; i += 256) {
    int r = i / 7, c = i - r * 7;
    int node = nodeBase + r;
    int cl = node < N_NODES ? node : (N_NODES - 1);
    zs[r][c] = x[cl * 7 + c] + aggr1[cl * 7 + c];
  }
  for (int i = tid; i < 7 * 64; i += 256) w1a_s[i >> 6][i & 63] = W1a[i];
  if (tid < 64) b1a_s[tid] = b1a[tid];
  for (int i = tid; i < 64 * 64; i += 256) {
    int k = i >> 6, n = i & 63;
    w1bt[n][k] = (_Float16)W1b[i];
  }
  __syncthreads();

  // t = relu(z @ W1a + b1a)  (K=7 -> scalar FMA)
  for (int i = tid; i < 128 * 64; i += 256) {
    int r = i >> 6, c = i & 63;
    float acc = b1a_s[c];
#pragma unroll
    for (int k = 0; k < 7; ++k) acc = fmaf(zs[r][k], w1a_s[k][c], acc);
    t16[r][c] = (_Float16)fmaxf(acc, 0.0f);
  }
  __syncthreads();

  // h1 = relu(t @ W1b + b1b) via WMMA (wave owns a 16-row tile)
  int lane = tid & 31, wave = tid >> 5;
  int row0 = wave * 16;
  v16h a0 = load_A_frag(&t16[0][0], row0, 0, lane);
  v16h a1 = load_A_frag(&t16[0][0], row0, 32, lane);
  int n = lane & 15, mo = (lane >> 4) * 8;
#pragma unroll
  for (int nc = 0; nc < 4; ++nc) {
    v8f acc = {};
    v16h b0 = load_B_fragT(&w1bt[0][0], 0,  nc * 16, lane);
    v16h b1 = load_B_fragT(&w1bt[0][0], 32, nc * 16, lane);
    acc = __builtin_amdgcn_wmma_f32_16x16x32_f16(false, a0, false, b0,
                                                 (short)0, acc, false, false);
    acc = __builtin_amdgcn_wmma_f32_16x16x32_f16(false, a1, false, b1,
                                                 (short)0, acc, false, false);
    float bias = b1b[nc * 16 + n];
#pragma unroll
    for (int r = 0; r < 8; ++r)
      outs[row0 + mo + r][nc * 16 + n] = fmaxf(acc[r] + bias, 0.0f);
  }
  __syncthreads();
  // coalesced b128 stores with uniform row bound
  int rows = N_NODES - nodeBase; if (rows > 128) rows = 128;
  for (int i = tid; i < rows * 16; i += 256) {
    int r = i >> 4, c4 = (i & 15) << 2;
    *(float4*)&h1[(nodeBase + r) * 64 + c4] = *(const float4*)&outs[r][c4];
  }
}

// ---- Layer 2 edge phase: 4 threads/edge, 16-dim chunks ---------------------
__global__ __launch_bounds__(256) void k_edge2(
    const float* __restrict__ edge_attr, const int* __restrict__ eidx,
    const float* __restrict__ h1, const float* __restrict__ We2,
    const float* __restrict__ be2, float* __restrict__ aggr2) {
  __shared__ float we[4][64];
  __shared__ float be[64];
  int tid = threadIdx.x;
  if (tid < 256) we[tid >> 6][tid & 63] = We2[tid];
  if (tid < 64)  be[tid] = be2[tid];
  __syncthreads();
  int t = blockIdx.x * 256 + tid;
  int e = t >> 2;
  if (e >= N_EDGES) return;
  int c0 = (t & 3) << 4;
  float4 ea = ((const float4*)edge_attr)[e];
  int src = eidx[e];
  int dst = eidx[N_EDGES + e];
  const float4* hrow = (const float4*)(h1 + src * 64 + c0);   // L2-resident
  float* arow = aggr2 + dst * 64 + c0;
  float hbuf[16];
#pragma unroll
  for (int q = 0; q < 4; ++q) {
    float4 hv = hrow[q];
    hbuf[4 * q + 0] = hv.x; hbuf[4 * q + 1] = hv.y;
    hbuf[4 * q + 2] = hv.z; hbuf[4 * q + 3] = hv.w;
  }
#pragma unroll
  for (int d = 0; d < 16; ++d) {
    int dim = c0 + d;
    float v = be[dim] + ea.x * we[0][dim] + ea.y * we[1][dim]
                      + ea.z * we[2][dim] + ea.w * we[3][dim] + hbuf[d];
    v = fmaxf(v, 0.0f);
    atomicAddF(arow + d, v);
  }
}

// ---- Layer 2 node MLP: h2 = relu( relu((h1+aggr2)@W2a+b2a) @ W2b + b2b ) ---
__global__ __launch_bounds__(256) void k_node_mlp2(
    const float* __restrict__ h1, const float* __restrict__ aggr2,
    const float* __restrict__ W2a, const float* __restrict__ b2a,
    const float* __restrict__ W2b, const float* __restrict__ b2b,
    float* __restrict__ h2) {
  __shared__ _Float16 z16[128][64];      // z tile, later reused for t tile
  __shared__ _Float16 wat[64][64];       // W2a transposed
  __shared__ _Float16 wbt[64][64];       // W2b transposed
  __shared__ float    outs[128][64];     // f32 result staging
  int tid = threadIdx.x;
  int nodeBase = blockIdx.x * 128;
  for (int i = tid; i < 128 * 64; i += 256) {
    int r = i >> 6, c = i & 63;
    int node = nodeBase + r;
    int cl = node < N_NODES ? node : (N_NODES - 1);
    z16[r][c] = (_Float16)(h1[cl * 64 + c] + aggr2[cl * 64 + c]);
  }
  for (int i = tid; i < 64 * 64; i += 256) {
    int k = i >> 6, n = i & 63;
    wat[n][k] = (_Float16)W2a[i];
    wbt[n][k] = (_Float16)W2b[i];
  }
  __syncthreads();

  int lane = tid & 31, wave = tid >> 5;
  int row0 = wave * 16;
  int n = lane & 15, mo = (lane >> 4) * 8;

  // GEMM1: t = relu(z @ W2a + b2a)
  v16h a0 = load_A_frag(&z16[0][0], row0, 0, lane);
  v16h a1 = load_A_frag(&z16[0][0], row0, 32, lane);
  v8f acc[4] = {};
#pragma unroll
  for (int nc = 0; nc < 4; ++nc) {
    v16h b0 = load_B_fragT(&wat[0][0], 0,  nc * 16, lane);
    v16h b1 = load_B_fragT(&wat[0][0], 32, nc * 16, lane);
    acc[nc] = __builtin_amdgcn_wmma_f32_16x16x32_f16(false, a0, false, b0,
                                                     (short)0, acc[nc], false, false);
    acc[nc] = __builtin_amdgcn_wmma_f32_16x16x32_f16(false, a1, false, b1,
                                                     (short)0, acc[nc], false, false);
  }
  __syncthreads();
  // write t back into z16 (each wave touches only its own 16 rows)
#pragma unroll
  for (int nc = 0; nc < 4; ++nc) {
    float bias = b2a[nc * 16 + n];
#pragma unroll
    for (int r = 0; r < 8; ++r)
      z16[row0 + mo + r][nc * 16 + n] = (_Float16)fmaxf(acc[nc][r] + bias, 0.0f);
  }
  __syncthreads();

  // GEMM2: h2 = relu(t @ W2b + b2b)
  a0 = load_A_frag(&z16[0][0], row0, 0, lane);
  a1 = load_A_frag(&z16[0][0], row0, 32, lane);
#pragma unroll
  for (int nc = 0; nc < 4; ++nc) {
    v8f acc2 = {};
    v16h b0 = load_B_fragT(&wbt[0][0], 0,  nc * 16, lane);
    v16h b1 = load_B_fragT(&wbt[0][0], 32, nc * 16, lane);
    acc2 = __builtin_amdgcn_wmma_f32_16x16x32_f16(false, a0, false, b0,
                                                  (short)0, acc2, false, false);
    acc2 = __builtin_amdgcn_wmma_f32_16x16x32_f16(false, a1, false, b1,
                                                  (short)0, acc2, false, false);
    float bias = b2b[nc * 16 + n];
#pragma unroll
    for (int r = 0; r < 8; ++r)
      outs[row0 + mo + r][nc * 16 + n] = fmaxf(acc2[r] + bias, 0.0f); // outer relu
  }
  __syncthreads();
  int rows = N_NODES - nodeBase; if (rows > 128) rows = 128;
  for (int i = tid; i < rows * 16; i += 256) {
    int r = i >> 4, c4 = (i & 15) << 2;
    *(float4*)&h2[(nodeBase + r) * 64 + c4] = *(const float4*)&outs[r][c4];
  }
}

// ---- Mean-pool (atomic accumulate) -----------------------------------------
__global__ __launch_bounds__(256) void k_pool(
    const float* __restrict__ h2, const int* __restrict__ batch,
    float* __restrict__ sums, float* __restrict__ cnt) {
  int t = blockIdx.x * 256 + threadIdx.x;
  int node = t >> 2;
  if (node >= N_NODES) return;
  int c0 = (t & 3) << 4;
  int g = batch[node];
  const float* hrow = h2 + node * 64 + c0;
  float* srow = sums + g * 64 + c0;
#pragma unroll
  for (int d = 0; d < 16; ++d) atomicAddF(srow + d, hrow[d]);
  if ((t & 3) == 0) atomicAddF(cnt + g, 1.0f);
}

// ---- Final FC: out = (sums/max(cnt,1)) @ Wfc + bfc -------------------------
__global__ __launch_bounds__(256) void k_fc(
    const float* __restrict__ sums, const float* __restrict__ cnt,
    const float* __restrict__ Wfc, const float* __restrict__ bfc,
    float* __restrict__ out) {
  int t = blockIdx.x * 256 + threadIdx.x;
  if (t >= N_GRAPHS * 12) return;
  int g = t / 12, j = t - g * 12;
  float inv = 1.0f / fmaxf(cnt[g], 1.0f);
  float acc = bfc[j];
#pragma unroll 8
  for (int k = 0; k < 64; ++k) acc = fmaf(sums[g * 64 + k] * inv, Wfc[k * 12 + j], acc);
  out[t] = acc;
}

// ---------------------------------------------------------------------------
extern "C" void kernel_launch(void* const* d_in, const int* in_sizes, int n_in,
                              void* d_out, int out_size, void* d_ws, size_t ws_size,
                              hipStream_t stream) {
  const float* x         = (const float*)d_in[0];
  const float* edge_attr = (const float*)d_in[1];
  const int*   eidx      = (const int*)  d_in[2];
  const int*   batch     = (const int*)  d_in[3];
  const float* We1 = (const float*)d_in[4];
  const float* be1 = (const float*)d_in[5];
  const float* W1a = (const float*)d_in[6];
  const float* b1a = (const float*)d_in[7];
  const float* W1b = (const float*)d_in[8];
  const float* b1b = (const float*)d_in[9];
  const float* We2 = (const float*)d_in[10];
  const float* be2 = (const float*)d_in[11];
  const float* W2a = (const float*)d_in[12];
  const float* b2a = (const float*)d_in[13];
  const float* W2b = (const float*)d_in[14];
  const float* b2b = (const float*)d_in[15];
  const float* Wfc = (const float*)d_in[16];
  const float* bfc = (const float*)d_in[17];
  float* out = (float*)d_out;

  // Workspace layout (floats). Zeroed region first (contiguous).
  float* ws    = (float*)d_ws;
  float* aggr1 = ws;                       // 350,000
  float* aggr2 = aggr1 + 350000;           // 3,200,000
  float* sums  = aggr2 + 3200000;          // 64,000
  float* cnt   = sums  + 64000;            // 1,024 (pad)
  float* h1    = cnt   + 1024;             // 3,200,000
  float* h2    = h1    + 3200000;          // 3,200,000
  const int ZERO_N = 350000 + 3200000 + 64000 + 1024;

  dim3 blk(256);
  k_zero<<<dim3((ZERO_N + 255) / 256), blk, 0, stream>>>(ws, ZERO_N);
  k_edge1<<<dim3((N_EDGES + 255) / 256), blk, 0, stream>>>(
      edge_attr, eidx, x, We1, be1, aggr1);
  k_node_mlp1<<<dim3((N_NODES + 127) / 128), blk, 0, stream>>>(
      x, aggr1, W1a, b1a, W1b, b1b, h1);
  k_edge2<<<dim3((N_EDGES * 4 + 255) / 256), blk, 0, stream>>>(
      edge_attr, eidx, h1, We2, be2, aggr2);
  k_node_mlp2<<<dim3((N_NODES + 127) / 128), blk, 0, stream>>>(
      h1, aggr2, W2a, b2a, W2b, b2b, h2);
  k_pool<<<dim3((N_NODES * 4 + 255) / 256), blk, 0, stream>>>(
      h2, batch, sums, cnt);
  k_fc<<<dim3((N_GRAPHS * 12 + 255) / 256), blk, 0, stream>>>(
      sums, cnt, Wfc, bfc, out);
}
